// FCOS_54966991454498
// MI455X (gfx1250) — compile-verified, matrix operand order checked by reading, not run on hardware
//
#include <hip/hip_runtime.h>

typedef __attribute__((ext_vector_type(16))) _Float16 v16h;
typedef __attribute__((ext_vector_type(8)))  _Float16 v8h;
typedef __attribute__((ext_vector_type(8)))  float    v8f;

#define CIN  256
#define KTOT 2304   // 256 * 9
#define NKB  72     // 2304 / 32

// ---------------------------------------------------------------------------
// Implicit-GEMM 3x3 SAME conv over pre-padded NHWC f16 activations
// act layout: [2, H+2, W+2, 256] with zeroed 1-pixel halo -> NO bounds logic.
//   out[n,co,p] = sum_k Wp[co,k] * im2col(act)[k,p] + bias[co]
// K order: k = (ky*3+kx)*256 + ci -> per 32-K-block: fixed tap, contiguous ci
//   == one contiguous 64B run in padded NHWC -> plain b128 loads.
// Tile: 64 (co) x 128 (pixels) per 256-thread block; 8 waves x 4 accumulators.
// Double-buffered LDS: one barrier per K-step; staging overlaps WMMA.
// GUARD=false: Mout == 256 and HW multiple of 128 (no tile guards at all).
// ---------------------------------------------------------------------------
template <bool GUARD>
__global__ __launch_bounds__(256) void conv3x3_wmma(
    const _Float16* __restrict__ act,   // [2, H+2, W+2, 256] f16 padded NHWC
    const _Float16* __restrict__ Wp,    // [Mout, 2304] f16 repacked
    const float*    __restrict__ bias,  // [Mout]
    float*          __restrict__ out,   // [2, Mout, H*W] f32 (NCHW)
    int H, int W, int Mout)
{
  __shared__ __align__(16) _Float16 As[2][64 * 32];    // 2 x 4 KB
  __shared__ __align__(16) _Float16 Bs[2][128 * 32];   // 2 x 8 KB

  const int HW    = H * W;
  const int PW    = W + 2;
  const int n     = blockIdx.z;
  const int pbase = blockIdx.x * 128;
  const int mbase = blockIdx.y * 64;
  const int tid   = threadIdx.x;
  const int lane  = tid & 31;
  const int wave  = tid >> 5;
  const int msub  = wave & 3;   // 0..3 -> 16-row slice of M
  const int nsub  = wave >> 2;  // 0..1 -> 64-pixel slice of N

  // ---- A-tile staging: one b128 per thread per K-step ----
  const int arow = tid >> 2;
  const int aseg = tid & 3;
  const int am   = mbase + arow;
  const bool aok = GUARD ? (am < Mout) : true;
  const _Float16* wrow = Wp + (size_t)(aok ? am : 0) * KTOT + aseg * 8;

  v8h zero8;
#pragma unroll
  for (int e = 0; e < 8; ++e) zero8[e] = (_Float16)0.f;

  // ---- B-tile staging: two b128 per thread per K-step, halo-padded source ---
  // slots tid and tid+256 cover 512 (pixel,seg) pairs = 128 px * 4 segs
  const _Float16* actn = act + (size_t)n * (H + 2) * PW * CIN;
  int bpix[2], bseg[2];
  const _Float16* bptr[2];   // pointer at padded (h+1, w+1) for this pixel/seg
#pragma unroll
  for (int j = 0; j < 2; ++j) {
    const int id = tid + j * 256;
    bpix[j] = id >> 2;
    bseg[j] = id & 3;
    int p = pbase + bpix[j];
    if (GUARD) p = p < HW ? p : 0;       // clamp: valid memory, column discarded
    const int h = p / W, w = p % W;
    bptr[j] = actn + ((size_t)(h + 1) * PW + (w + 1)) * CIN + bseg[j] * 8;
  }

  // staging: global -> regs (no bounds logic; halo supplies zeros)
  auto load_tile = [&](int kb, v8h& ar, v8h& br0, v8h& br1) {
    ar = *(const v8h*)(wrow + kb * 32);
    if (GUARD) ar = aok ? ar : zero8;
    const int tap = kb >> 3;
    const int cib = (kb & 7) << 5;
    const int doff = ((tap / 3 - 1) * PW + (tap % 3 - 1)) * CIN + cib; // uniform
    br0 = *(const v8h*)(bptr[0] + doff);
    br1 = *(const v8h*)(bptr[1] + doff);
  };
  // staging: regs -> LDS buffer
  auto store_tile = [&](int buf, v8h ar, v8h br0, v8h br1) {
    *(v8h*)&As[buf][arow * 32 + aseg * 8] = ar;
    *(v8h*)&Bs[buf][bpix[0] * 32 + bseg[0] * 8] = br0;
    *(v8h*)&Bs[buf][bpix[1] * 32 + bseg[1] * 8] = br1;
  };

  v8f acc[4] = {{}, {}, {}, {}};

  // ---- per-lane fragment offsets (16-bit 16x32 WMMA layout) ----
  const int g = lane >> 4;
  const int r = lane & 15;
  const int a_off = (msub * 16 + r) * 32 + g * 8;
  int b_off[4];
#pragma unroll
  for (int c = 0; c < 4; ++c) b_off[c] = (nsub * 64 + c * 16 + r) * 32 + g * 8;

  // prologue: stage K-block 0 into buffer 0
  {
    v8h ar, br0, br1;
    load_tile(0, ar, br0, br1);
    store_tile(0, ar, br0, br1);
  }
  __syncthreads();

  for (int kb = 0; kb < NKB; ++kb) {
    const int cur = kb & 1;
    v8h na, nb0, nb1;
    if (kb + 1 < NKB) load_tile(kb + 1, na, nb0, nb1);

    v8h a0 = *(const v8h*)&As[cur][a_off];
    v8h a1 = *(const v8h*)&As[cur][a_off + 16];
    v16h a;
#pragma unroll
    for (int e = 0; e < 8; ++e) { a[e] = a0[e]; a[e + 8] = a1[e]; }
#pragma unroll
    for (int c = 0; c < 4; ++c) {
      v8h blo = *(const v8h*)&Bs[cur][b_off[c]];
      v8h bhi = *(const v8h*)&Bs[cur][b_off[c] + 16];
      v16h b;
#pragma unroll
      for (int e = 0; e < 8; ++e) { b[e] = blo[e]; b[e + 8] = bhi[e]; }
      acc[c] = __builtin_amdgcn_wmma_f32_16x16x32_f16(false, a, false, b, (short)0,
                                                      acc[c], false, false);
    }

    if (kb + 1 < NKB) {
      store_tile(cur ^ 1, na, nb0, nb1);   // fill other buffer
      __syncthreads();                     // publish; also fences reads of cur
    }
  }

  // C/D layout: lanes 0-15 -> M=e, N=lane; lanes 16-31 -> M=8+e, N=lane-16
  const int mrow = mbase + msub * 16 + (g ? 8 : 0);
#pragma unroll
  for (int e = 0; e < 8; ++e) {
    const int co = mrow + e;
    if (!GUARD || co < Mout) {
      const float bv = bias[co];
      float* orow = out + ((size_t)n * Mout + co) * HW;
#pragma unroll
      for (int c = 0; c < 4; ++c) {
        const int p = pbase + nsub * 64 + c * 16 + r;
        if (!GUARD || p < HW) orow[p] = acc[c][e] + bv;
      }
    }
  }
}

// ---------------------------------------------------------------------------
// vectorized zero-fill (for padded-activation halos)
// ---------------------------------------------------------------------------
__global__ void zero_f16v(v8h* __restrict__ y, long n8)
{
  v8h z;
#pragma unroll
  for (int e = 0; e < 8; ++e) z[e] = (_Float16)0.f;
  for (long i = (long)blockIdx.x * 256 + threadIdx.x; i < n8; i += (long)gridDim.x * 256)
    y[i] = z;
}

// ---------------------------------------------------------------------------
// GroupNorm stats over NCHW f32 conv output: one block per (n, group).
// ---------------------------------------------------------------------------
__global__ __launch_bounds__(256) void gn_stats(
    const float* __restrict__ x, float* __restrict__ stats, int HW)
{
  __shared__ float rs[256], rq[256];
  const int bi  = blockIdx.x;           // 0..31 = n*16 + group
  const int n   = bi >> 4;
  const int grp = bi & 15;
  const float* base = x + ((size_t)n * 256 + grp * 16) * HW;
  const long cnt = 16L * HW;
  float s = 0.f, q = 0.f;
  for (long i = threadIdx.x; i < cnt; i += 256) {
    float v = base[i]; s += v; q += v * v;
  }
  rs[threadIdx.x] = s; rq[threadIdx.x] = q;
  __syncthreads();
  for (int st = 128; st > 0; st >>= 1) {
    if ((int)threadIdx.x < st) {
      rs[threadIdx.x] += rs[threadIdx.x + st];
      rq[threadIdx.x] += rq[threadIdx.x + st];
    }
    __syncthreads();
  }
  if (threadIdx.x == 0) {
    float mean = rs[0] / (float)cnt;
    float var  = rq[0] / (float)cnt - mean * mean;
    stats[bi * 2]     = mean;
    stats[bi * 2 + 1] = rsqrtf(var + 1e-5f);
  }
}

// NCHW f32 -> padded NHWC f16 with GN scale/bias + ReLU (coalesced writes)
__global__ void gn_apply_relu_nhwc_pad(
    const float* __restrict__ x, const float* __restrict__ stats,
    const float* __restrict__ sc, const float* __restrict__ bi,
    _Float16* __restrict__ y, int H, int W)
{
  const int HW = H * W, PW = W + 2;
  const long total = 2L * HW * 256;
  for (long idx = (long)blockIdx.x * 256 + threadIdx.x; idx < total;
       idx += (long)gridDim.x * 256) {
    const int c = (int)(idx & 255);
    const long pn = idx >> 8;              // n*HW + p
    const int p = (int)(pn % HW);
    const int n = (int)(pn / HW);
    const int h = p / W, w = p % W;
    const int grp = c >> 4;
    const float mean = stats[(n * 16 + grp) * 2];
    const float rstd = stats[(n * 16 + grp) * 2 + 1];
    const float v = (x[((size_t)n * 256 + c) * HW + p] - mean) * rstd * sc[c] + bi[c];
    y[(((size_t)n * (H + 2) + h + 1) * PW + (w + 1)) * 256 + c] = (_Float16)fmaxf(v, 0.f);
  }
}

// NCHW f32 -> padded NHWC f16 plain convert (input features)
__global__ void nchw_to_nhwc_pad_f16(const float* __restrict__ x, _Float16* __restrict__ y,
                                     int H, int W)
{
  const int HW = H * W, PW = W + 2;
  const long total = 2L * HW * 256;
  for (long idx = (long)blockIdx.x * 256 + threadIdx.x; idx < total;
       idx += (long)gridDim.x * 256) {
    const int c = (int)(idx & 255);
    const long pn = idx >> 8;
    const int p = (int)(pn % HW);
    const int n = (int)(pn / HW);
    const int h = p / W, w = p % W;
    y[(((size_t)n * (H + 2) + h + 1) * PW + (w + 1)) * 256 + c] =
        (_Float16)x[((size_t)n * 256 + c) * HW + p];
  }
}

// Repack W[o,i,3,3] f32 -> Wp[o][(ky*3+kx)*256 + i] f16
__global__ void repack_w(const float* __restrict__ w, _Float16* __restrict__ wp, int Mout)
{
  const long total = (long)Mout * KTOT;
  for (long idx = (long)blockIdx.x * 256 + threadIdx.x; idx < total; idx += (long)gridDim.x * 256) {
    const int o = (int)(idx / KTOT);
    const int k = (int)(idx % KTOT);
    const int t = k >> 8;
    const int i = k & 255;
    wp[idx] = (_Float16)w[((size_t)o * 256 + i) * 9 + t];
  }
}

// box(4) + ctr(1) fused into padded M=16 GEMM
__global__ void repack_boxctr(const float* __restrict__ box_w, const float* __restrict__ ctr_w,
                              _Float16* __restrict__ wp)
{
  const long total = 16L * KTOT;
  for (long idx = (long)blockIdx.x * 256 + threadIdx.x; idx < total; idx += (long)gridDim.x * 256) {
    const int o = (int)(idx / KTOT);
    const int k = (int)(idx % KTOT);
    const int t = k >> 8;
    const int i = k & 255;
    float v = 0.f;
    if (o < 4)       v = box_w[((size_t)o * 256 + i) * 9 + t];
    else if (o == 4) v = ctr_w[(size_t)i * 9 + t];
    wp[idx] = (_Float16)v;
  }
}

__global__ void make_bias16(const float* __restrict__ box_b, const float* __restrict__ ctr_b,
                            float* __restrict__ b16)
{
  const int i = threadIdx.x;
  if (i < 16) b16[i] = (i < 4) ? box_b[i] : ((i == 4) ? ctr_b[0] : 0.f);
}

// [2,80,HW] -> cls[n, pbase+p, c]
__global__ void scatter_cls(const float* __restrict__ t, float* __restrict__ o,
                            int HW, int pbase, int P)
{
  const long total = 2L * 80 * HW;
  for (long idx = (long)blockIdx.x * 256 + threadIdx.x; idx < total; idx += (long)gridDim.x * 256) {
    const int n = (int)(idx / (80L * HW));
    const int c = (int)((idx / HW) % 80);
    const int p = (int)(idx % HW);
    o[((size_t)n * P + pbase + p) * 80 + c] = t[idx];
  }
}

// [2,16,HW]: rows 0-3 = box (ReLU), row 4 = ctr
__global__ void scatter_regctr(const float* __restrict__ t, float* __restrict__ oreg,
                               float* __restrict__ octr, int HW, int pbase, int P)
{
  const long total = 2L * 5 * HW;
  for (long idx = (long)blockIdx.x * 256 + threadIdx.x; idx < total; idx += (long)gridDim.x * 256) {
    const int n = (int)(idx / (5L * HW));
    const int c = (int)((idx / HW) % 5);
    const int p = (int)(idx % HW);
    const float v = t[((size_t)n * 16 + c) * HW + p];
    if (c < 4) oreg[((size_t)n * P + pbase + p) * 4 + c] = fmaxf(v, 0.f);
    else       octr[(size_t)n * P + pbase + p] = v;
  }
}

// ---------------------------------------------------------------------------

static inline int gsz(long total) {
  long b = (total + 255) / 256;
  return (int)(b > 4096 ? 4096 : (b < 1 ? 1 : b));
}

extern "C" void kernel_launch(void* const* d_in, const int* in_sizes, int n_in,
                              void* d_out, int out_size, void* d_ws, size_t ws_size,
                              hipStream_t stream) {
  (void)in_sizes; (void)n_in; (void)out_size; (void)ws_size;

  const float* feat[5] = {(const float*)d_in[0], (const float*)d_in[1], (const float*)d_in[2],
                          (const float*)d_in[3], (const float*)d_in[4]};
  const float* cls_w0   = (const float*)d_in[5];
  const float* cls_b0   = (const float*)d_in[6];
  const float* cls_gs0  = (const float*)d_in[7];
  const float* cls_gb0  = (const float*)d_in[8];
  const float* cls_w1   = (const float*)d_in[9];
  const float* cls_b1   = (const float*)d_in[10];
  const float* cls_gs1  = (const float*)d_in[11];
  const float* cls_gb1  = (const float*)d_in[12];
  const float* reg_w0   = (const float*)d_in[13];
  const float* reg_b0   = (const float*)d_in[14];
  const float* reg_gs0  = (const float*)d_in[15];
  const float* reg_gb0  = (const float*)d_in[16];
  const float* reg_w1   = (const float*)d_in[17];
  const float* reg_b1   = (const float*)d_in[18];
  const float* reg_gs1  = (const float*)d_in[19];
  const float* reg_gb1  = (const float*)d_in[20];
  const float* clso_w   = (const float*)d_in[21];
  const float* clso_b   = (const float*)d_in[22];
  const float* box_w    = (const float*)d_in[23];
  const float* box_b    = (const float*)d_in[24];
  const float* ctr_w    = (const float*)d_in[25];
  const float* ctr_b    = (const float*)d_in[26];

  const int P = 21824;
  float* out_cls = (float*)d_out;
  float* out_reg = out_cls + 2L * P * 80;
  float* out_ctr = out_reg + 2L * P * 4;

  // workspace carve-out (256B aligned)
  char* ws = (char*)d_ws;
  size_t off = 0;
  auto carve = [&](size_t bytes) -> char* {
    char* r = ws + off;
    off = (off + bytes + 255) & ~(size_t)255;
    return r;
  };
  _Float16* wp_cls0 = (_Float16*)carve(256L * KTOT * 2);
  _Float16* wp_cls1 = (_Float16*)carve(256L * KTOT * 2);
  _Float16* wp_reg0 = (_Float16*)carve(256L * KTOT * 2);
  _Float16* wp_reg1 = (_Float16*)carve(256L * KTOT * 2);
  _Float16* wp_cout = (_Float16*)carve(80L * KTOT * 2);
  _Float16* wp_bc   = (_Float16*)carve(16L * KTOT * 2);
  float*    bias_bc = (float*)carve(16 * 4);
  float*    stats   = (float*)carve(32 * 2 * 4);
  const long MAXPE  = 2L * 130 * 130 * 256;         // padded level-0 elements
  const long MAXE   = 2L * 256 * 16384;
  _Float16* xf16    = (_Float16*)carve(MAXPE * 2);  // padded NHWC input
  _Float16* actb    = (_Float16*)carve(MAXPE * 2);  // padded NHWC activations
  float*    tmp     = (float*)carve(MAXE * 4);      // NCHW conv f32 output

  // one-time weight repack (f32 -> f16, K-major (tap,ci))
  repack_w<<<gsz(256L * KTOT), 256, 0, stream>>>(cls_w0, wp_cls0, 256);
  repack_w<<<gsz(256L * KTOT), 256, 0, stream>>>(cls_w1, wp_cls1, 256);
  repack_w<<<gsz(256L * KTOT), 256, 0, stream>>>(reg_w0, wp_reg0, 256);
  repack_w<<<gsz(256L * KTOT), 256, 0, stream>>>(reg_w1, wp_reg1, 256);
  repack_w<<<gsz(80L * KTOT), 256, 0, stream>>>(clso_w, wp_cout, 80);
  repack_boxctr<<<gsz(16L * KTOT), 256, 0, stream>>>(box_w, ctr_w, wp_bc);
  make_bias16<<<1, 16, 0, stream>>>(box_b, ctr_b, bias_bc);

  const int Hs[5] = {128, 64, 32, 16, 8};
  int pbase = 0;
  for (int lvl = 0; lvl < 5; ++lvl) {
    const int H = Hs[lvl], W = Hs[lvl], HW = H * W;
    const long nelem = 2L * 256 * HW;
    const long npad8 = 2L * (H + 2) * (W + 2) * 256 / 8;   // v8h count

    // zero halos (interior is overwritten by the NHWC writers)
    zero_f16v<<<gsz(npad8), 256, 0, stream>>>((v8h*)xf16, npad8);
    zero_f16v<<<gsz(npad8), 256, 0, stream>>>((v8h*)actb, npad8);
    nchw_to_nhwc_pad_f16<<<gsz(nelem), 256, 0, stream>>>(feat[lvl], xf16, H, W);

    const int nx = (HW + 127) / 128;
    const bool full = (HW % 128) == 0;           // all levels except 8x8
    const dim3 cg256(nx, 4, 2);  // Mout=256 -> 4 M-blocks
    const dim3 cg80(nx, 2, 2);   // Mout=80  -> 2 M-blocks
    const dim3 cg16(nx, 1, 2);   // Mout=16  -> 1 M-block

    auto conv256 = [&](const _Float16* a, const _Float16* w, const float* b, float* o) {
      if (full) conv3x3_wmma<false><<<cg256, 256, 0, stream>>>(a, w, b, o, H, W, 256);
      else      conv3x3_wmma<true ><<<cg256, 256, 0, stream>>>(a, w, b, o, H, W, 256);
    };

    // ---- classification tower ----
    conv256(xf16, wp_cls0, cls_b0, tmp);
    gn_stats<<<32, 256, 0, stream>>>(tmp, stats, HW);
    gn_apply_relu_nhwc_pad<<<gsz(nelem), 256, 0, stream>>>(tmp, stats, cls_gs0, cls_gb0, actb, H, W);
    conv256(actb, wp_cls1, cls_b1, tmp);
    gn_stats<<<32, 256, 0, stream>>>(tmp, stats, HW);
    gn_apply_relu_nhwc_pad<<<gsz(nelem), 256, 0, stream>>>(tmp, stats, cls_gs1, cls_gb1, actb, H, W);
    conv3x3_wmma<true><<<cg80, 256, 0, stream>>>(actb, wp_cout, clso_b, tmp, H, W, 80);
    scatter_cls<<<gsz(2L * 80 * HW), 256, 0, stream>>>(tmp, out_cls, HW, pbase, P);

    // ---- regression tower ----
    conv256(xf16, wp_reg0, reg_b0, tmp);
    gn_stats<<<32, 256, 0, stream>>>(tmp, stats, HW);
    gn_apply_relu_nhwc_pad<<<gsz(nelem), 256, 0, stream>>>(tmp, stats, reg_gs0, reg_gb0, actb, H, W);
    conv256(actb, wp_reg1, reg_b1, tmp);
    gn_stats<<<32, 256, 0, stream>>>(tmp, stats, HW);
    gn_apply_relu_nhwc_pad<<<gsz(nelem), 256, 0, stream>>>(tmp, stats, reg_gs1, reg_gb1, actb, H, W);
    conv3x3_wmma<true><<<cg16, 256, 0, stream>>>(actb, wp_bc, bias_bc, tmp, H, W, 16);
    scatter_regctr<<<gsz(2L * 5 * HW), 256, 0, stream>>>(tmp, out_reg, out_ctr, HW, pbase, P);

    pbase += HW;
  }
}
